// GraphNN_43258910605712
// MI455X (gfx1250) — matrix operand order repaired and checked by reading, not compiled, not run on hardware
//
#include <hip/hip_runtime.h>
#include <math.h>

// ---------------------------------------------------------------------------
// CDNA5 (gfx1250) wave32 + WMMA bf16 GCN pipeline.
// GEMMs run on v_wmma_f32_16x16x32_bf16 with a bf16-hi/lo split (3 WMMAs per
// k-tile) for near-fp32 accuracy. Edge aggregation is coalesced loads +
// global_atomic_add_f32, which is the memory-bound part (~5 GB @ 23.3 TB/s).
// ---------------------------------------------------------------------------

typedef __attribute__((ext_vector_type(16))) __bf16 v16bf;
typedef __attribute__((ext_vector_type(8)))  __bf16 v8bf;
typedef __attribute__((ext_vector_type(8)))  float  v8f;

union Frag16 { v16bf v; v8bf h[2]; };

#define TPB 256

// ---------------- degree / norm ----------------
__global__ __launch_bounds__(TPB) void deg_init_kernel(float* __restrict__ deg, int n) {
    int i = blockIdx.x * blockDim.x + threadIdx.x;
    if (i < n) deg[i] = 1.0f;   // self-loop contributes 1 at every node
}

__global__ __launch_bounds__(TPB) void deg_edge_kernel(const int* __restrict__ dst,
                                                       float* __restrict__ deg, int E) {
    int e = blockIdx.x * blockDim.x + threadIdx.x;
    if (e < E) atomicAdd(&deg[dst[e]], 1.0f);
}

__global__ __launch_bounds__(TPB) void dinv_kernel(float* __restrict__ deg, int n) {
    int i = blockIdx.x * blockDim.x + threadIdx.x;
    if (i < n) deg[i] = rsqrtf(deg[i]);   // deg >= 1 always
}

// ---------------- fp32 -> bf16 hi/lo split ----------------
__global__ __launch_bounds__(TPB) void split_kernel(const float* __restrict__ x,
                                                    __bf16* __restrict__ hi,
                                                    __bf16* __restrict__ lo, int n) {
    int i = blockIdx.x * blockDim.x + threadIdx.x;
    if (i >= n) return;
    float v = x[i];
    __bf16 h = (__bf16)v;
    hi[i] = h;
    lo[i] = (__bf16)(v - (float)h);
}

// ---------------- pack W into B-fragment-ready layout ----------------
// Layout: Pack[(((nt*ktiles + kt)*32 + lane)*16 + j)] = W[kt*32 + (j<8 ? g*8 + j : 16 + g*8 + j-8)][nt*16 + c]
// where c = lane&15, g = lane>>4  (matches 16-bit 32x16 B VGPR layout, wave32).
__global__ __launch_bounds__(TPB) void pack_w_kernel(const float* __restrict__ W,
                                                     __bf16* __restrict__ Ph,
                                                     __bf16* __restrict__ Pl,
                                                     int Din, int Dout) {
    int t = blockIdx.x * blockDim.x + threadIdx.x;
    if (t >= Din * Dout) return;
    const int ktiles = Din >> 5;
    int j    = t & 15;
    int lane = (t >> 4) & 31;
    int rest = t >> 9;
    int kt   = rest % ktiles;
    int nt   = rest / ktiles;
    int c = lane & 15, g = lane >> 4;
    int K   = kt * 32 + (j < 8 ? g * 8 + j : 16 + g * 8 + (j - 8));
    int col = nt * 16 + c;
    float w = W[(size_t)K * Dout + col];
    __bf16 h = (__bf16)w;
    Ph[t] = h;
    Pl[t] = (__bf16)(w - (float)h);
}

// ---------------- WMMA GEMM: C[N x Dout] = A[N x Din] * W ----------------
// One wave per 16x16 output tile; bf16x3 (hi*hi + hi*lo + lo*hi), f32 accum.
__global__ __launch_bounds__(TPB)
void gemm_wmma_bf16x3_kernel(const __bf16* __restrict__ Ahi, const __bf16* __restrict__ Alo,
                             const __bf16* __restrict__ Bph, const __bf16* __restrict__ Bpl,
                             float* __restrict__ C, int Nrows, int Din, int Dout) {
    const int lane = threadIdx.x & 31;
    const int tile = blockIdx.x * (blockDim.x >> 5) + (threadIdx.x >> 5);
    const int ntiles = Dout >> 4;
    const int ktiles = Din >> 5;
    const int mt = tile / ntiles;
    const int nt = tile % ntiles;
    if (mt >= (Nrows >> 4)) return;

    const int m = lane & 15;       // A row within tile
    const int g = lane >> 4;       // lane half selects K sub-range
    const __bf16* arow_hi = Ahi + (size_t)(mt * 16 + m) * Din;
    const __bf16* arow_lo = Alo + (size_t)(mt * 16 + m) * Din;

    v8f acc = {};
    for (int kt = 0; kt < ktiles; ++kt) {
        const int k0 = kt * 32;
        Frag16 ah, al, bh, bl;
        // A 16x32 bf16 fragment: two contiguous 16B chunks per lane
        ah.h[0] = *(const v8bf*)(arow_hi + k0 + g * 8);
        ah.h[1] = *(const v8bf*)(arow_hi + k0 + 16 + g * 8);
        al.h[0] = *(const v8bf*)(arow_lo + k0 + g * 8);
        al.h[1] = *(const v8bf*)(arow_lo + k0 + 16 + g * 8);
        // B 32x16 bf16 fragment: pre-packed, 32 contiguous bytes per lane
        const size_t boff = ((size_t)(nt * ktiles + kt) * 32 + lane) * 16;
        bh.h[0] = *(const v8bf*)(Bph + boff);
        bh.h[1] = *(const v8bf*)(Bph + boff + 8);
        bl.h[0] = *(const v8bf*)(Bpl + boff);
        bl.h[1] = *(const v8bf*)(Bpl + boff + 8);

        acc = __builtin_amdgcn_wmma_f32_16x16x32_bf16(false, ah.v, false, bh.v,
                                                      (short)0, acc, false, false);
        acc = __builtin_amdgcn_wmma_f32_16x16x32_bf16(false, ah.v, false, bl.v,
                                                      (short)0, acc, false, false);
        acc = __builtin_amdgcn_wmma_f32_16x16x32_bf16(false, al.v, false, bh.v,
                                                      (short)0, acc, false, false);
    }
    // C/D layout: lane col = lane&15, rows = g*8 + r in VGPR r
    float* crow = C + (size_t)(mt * 16 + g * 8) * Dout + nt * 16 + (lane & 15);
#pragma unroll
    for (int r = 0; r < 8; ++r)
        crow[(size_t)r * Dout] = acc[r];
}

// ---------------- out[i][d] = h[i][d]*dinv[i]^2 + b[d]  (self loop + bias) --
__global__ __launch_bounds__(TPB) void selfloop_bias_kernel(const float* __restrict__ h,
                                                            const float* __restrict__ dinv,
                                                            const float* __restrict__ b,
                                                            float* __restrict__ out,
                                                            int n, int Dout, int dshift) {
    int t = blockIdx.x * blockDim.x + threadIdx.x;
    int i = t >> dshift;
    if (i >= n) return;
    int d = t & (Dout - 1);
    float di = dinv[i];
    out[t] = h[t] * di * di + b[d];
}

// ---------------- edge scatter: out[dst] += h[src] * dinv[src]*dinv[dst] ----
// One wave per edge; each lane owns PER contiguous floats of the feature row.
template <int PER>
__global__ __launch_bounds__(TPB)
void edge_scatter_kernel(const float* __restrict__ h, const float* __restrict__ dinv,
                         const int* __restrict__ src, const int* __restrict__ dst,
                         float* __restrict__ out, int E, int D) {
    int wid = (int)((blockIdx.x * blockDim.x + threadIdx.x) >> 5);
    if (wid >= E) return;
    int lane = threadIdx.x & 31;
    int s = src[wid], d = dst[wid];
    float w = dinv[s] * dinv[d];
    const float* hp = h + (size_t)s * D + lane * PER;
    float*       op = out + (size_t)d * D + lane * PER;
    float vals[PER];
#pragma unroll
    for (int i = 0; i < PER; ++i) vals[i] = hp[i];
#pragma unroll
    for (int i = 0; i < PER; ++i) atomicAdd(op + i, vals[i] * w);
}

// ---------------------------------------------------------------------------
extern "C" void kernel_launch(void* const* d_in, const int* in_sizes, int n_in,
                              void* d_out, int out_size, void* d_ws, size_t ws_size,
                              hipStream_t stream) {
    const float* emb = (const float*)d_in[0];   // [N, 128]
    const float* W1  = (const float*)d_in[1];   // [128, 256]
    const float* b1  = (const float*)d_in[2];   // [256]
    const float* W2  = (const float*)d_in[3];   // [256, 128]
    const float* b2  = (const float*)d_in[4];   // [128]
    const int*   ei  = (const int*)d_in[5];     // [2, E] flat

    const int N  = in_sizes[0] / 128;           // 100000
    const int E  = in_sizes[5] / 2;             // 1600000
    const int D1 = 128, D2 = 256;
    const int* srcIdx = ei;
    const int* dstIdx = ei + E;

    // ---- workspace bump allocator (256B aligned) ----
    char* ws = (char*)d_ws;
    size_t off = 0;
    auto alloc = [&](size_t bytes) -> void* {
        void* p = ws + off;
        off += (bytes + 255) & ~(size_t)255;
        return p;
    };
    float*  dinv  = (float*)alloc((size_t)N * 4);                 // deg -> dinv in place
    __bf16* W1ph  = (__bf16*)alloc((size_t)D1 * D2 * 2);
    __bf16* W1pl  = (__bf16*)alloc((size_t)D1 * D2 * 2);
    __bf16* W2ph  = (__bf16*)alloc((size_t)D2 * D1 * 2);
    __bf16* W2pl  = (__bf16*)alloc((size_t)D2 * D1 * 2);
    __bf16* inHi  = (__bf16*)alloc((size_t)N * D2 * 2);           // reused both layers
    __bf16* inLo  = (__bf16*)alloc((size_t)N * D2 * 2);
    float*  H     = (float*)alloc((size_t)N * D2 * 4);            // pre-aggregation GEMM out
    float*  out1  = (float*)alloc((size_t)N * D2 * 4);            // layer-1 output

    auto blocks = [](long long n) { return (unsigned)((n + TPB - 1) / TPB); };

    // degrees (with self-loops) -> dinv = deg^{-1/2}
    deg_init_kernel<<<blocks(N), TPB, 0, stream>>>(dinv, N);
    deg_edge_kernel<<<blocks(E), TPB, 0, stream>>>(dstIdx, dinv, E);
    dinv_kernel<<<blocks(N), TPB, 0, stream>>>(dinv, N);

    // pack weights into B-fragment layout (hi/lo bf16)
    pack_w_kernel<<<blocks((long long)D1 * D2), TPB, 0, stream>>>(W1, W1ph, W1pl, D1, D2);
    pack_w_kernel<<<blocks((long long)D2 * D1), TPB, 0, stream>>>(W2, W2ph, W2pl, D2, D1);

    // ---------------- layer 1 ----------------
    split_kernel<<<blocks((long long)N * D1), TPB, 0, stream>>>(emb, inHi, inLo, N * D1);

    long long tiles1 = (long long)(N / 16) * (D2 / 16);
    gemm_wmma_bf16x3_kernel<<<(unsigned)((tiles1 + 7) / 8), TPB, 0, stream>>>(
        inHi, inLo, W1ph, W1pl, H, N, D1, D2);

    selfloop_bias_kernel<<<blocks((long long)N * D2), TPB, 0, stream>>>(
        H, dinv, b1, out1, N, D2, 8);
    edge_scatter_kernel<8><<<(unsigned)((E + 7) / 8), TPB, 0, stream>>>(
        H, dinv, srcIdx, dstIdx, out1, E, D2);

    // ---------------- layer 2 ----------------
    split_kernel<<<blocks((long long)N * D2), TPB, 0, stream>>>(out1, inHi, inLo, N * D2);

    long long tiles2 = (long long)(N / 16) * (D1 / 16);
    gemm_wmma_bf16x3_kernel<<<(unsigned)((tiles2 + 7) / 8), TPB, 0, stream>>>(
        inHi, inLo, W2ph, W2pl, H, N, D2, D1);

    float* outF = (float*)d_out;
    selfloop_bias_kernel<<<blocks((long long)N * D1), TPB, 0, stream>>>(
        H, dinv, b2, outF, N, D1, 7);
    edge_scatter_kernel<4><<<(unsigned)((E + 7) / 8), TPB, 0, stream>>>(
        H, dinv, srcIdx, dstIdx, outF, E, D1);
}
